// DualRouteInteractionBlock_22539988369669
// MI455X (gfx1250) — matrix operand order repaired and checked by reading, not compiled, not run on hardware
//
#include <hip/hip_runtime.h>
#include <hip/hip_bf16.h>
#include <math.h>

// Problem constants (from reference)
#define BB   8
#define VV   321
#define SS   24
#define DM   256
#define HH   8
#define DHH  32
#define GG   10
#define KK   33      // ceil(V/G)
#define DFFN 1024

typedef __attribute__((ext_vector_type(16))) __bf16 v16bf;
typedef __attribute__((ext_vector_type(8)))  float  v8f;

__device__ __forceinline__ float gelu_exact(float x) {
  return 0.5f * x * (1.0f + erff(x * 0.70710678118654752f));
}

// -----------------------------------------------------------------------------
// Weight pre-swizzle: W (Kd x N, fp32 row-major) -> Wp bf16, fragment-friendly:
//   Wp[((k/32)*N + n)*32 + (k%32)] = bf16(W[k*N + n])
// A lane's WMMA-B fragment is then Wp + ((kt*N+n)*32 + 16*half), 16 contiguous
// bf16 = 32 bytes = two aligned b128 loads.
// -----------------------------------------------------------------------------
__global__ void wconv_kernel(const float* __restrict__ W, __bf16* __restrict__ Wp,
                             int Kd, int N) {
  size_t i = (size_t)blockIdx.x * blockDim.x + threadIdx.x;
  size_t total = (size_t)Kd * N;
  if (i >= total) return;
  int n = (int)(i % N);
  int k = (int)(i / N);
  Wp[(((size_t)(k >> 5) * N + n) << 5) + (k & 31)] = (__bf16)W[i];
}

// -----------------------------------------------------------------------------
// GEMM: C[M,N] = act(A[M,Kd] @ W[Kd,N] + bias[N]) (+ residual[M,N])
// A fp32 (converted in-register), W pre-swizzled bf16, WMMA 16x16x32 f32-acc.
// Block = 256 threads (8 waves). Block tile 128(M) x 64(N); wave tile 16x64.
// Kd multiple of 32, N multiple of 64 (true here: 256/1024).
// -----------------------------------------------------------------------------
__global__ void gemm_bf16_wmma(const float* __restrict__ A,
                               const __bf16* __restrict__ Wp,
                               const float* __restrict__ bias,
                               const float* __restrict__ residual,
                               float* __restrict__ C,
                               int M, int Kd, int N, int act) {
  const int lane  = threadIdx.x & 31;
  const int wave  = threadIdx.x >> 5;
  const int hlf   = lane >> 4;      // which 16-lane half
  const int l16   = lane & 15;
  const int mBase = blockIdx.y * 128 + wave * 16;
  const int nBase = blockIdx.x * 64;

  v8f acc[4];
#pragma unroll
  for (int t = 0; t < 4; ++t)
#pragma unroll
    for (int r = 0; r < 8; ++r) acc[t][r] = 0.0f;

  int mrow = mBase + l16;
  if (mrow >= M) mrow = M - 1;                  // clamp loads; stores guarded
  const float* __restrict__ arow = A + (size_t)mrow * Kd;

  // Per-lane base pointers for the 4 B fragments (advance by N*32 per K-tile)
  const __bf16* bptr[4];
#pragma unroll
  for (int t = 0; t < 4; ++t)
    bptr[t] = Wp + (((size_t)(nBase + 16 * t + l16)) << 5) + 16 * hlf;

  const int ktiles = Kd >> 5;
  for (int kt = 0; kt < ktiles; ++kt) {
    const int kb = kt << 5;
    // ---- wide loads first: 4x float4 for A, 2x 16-byte x4 for B ----
    const float4* a0 = (const float4*)(arow + kb + 16 * hlf);
    const float4* a1 = (const float4*)(arow + kb + 16 * hlf + 8);
    float4 p0 = a0[0], p1 = a0[1];
    float4 p2 = a1[0], p3 = a1[1];

    v16bf bfrag[4];
#pragma unroll
    for (int t = 0; t < 4; ++t)
      bfrag[t] = *(const v16bf*)(bptr[t] + (size_t)kt * N * 32);

    // ---- convert A run to bf16 (packed cvt) ----
    v16bf af;
    af[0]  = (__bf16)p0.x; af[1]  = (__bf16)p0.y;
    af[2]  = (__bf16)p0.z; af[3]  = (__bf16)p0.w;
    af[4]  = (__bf16)p1.x; af[5]  = (__bf16)p1.y;
    af[6]  = (__bf16)p1.z; af[7]  = (__bf16)p1.w;
    af[8]  = (__bf16)p2.x; af[9]  = (__bf16)p2.y;
    af[10] = (__bf16)p2.z; af[11] = (__bf16)p2.w;
    af[12] = (__bf16)p3.x; af[13] = (__bf16)p3.y;
    af[14] = (__bf16)p3.z; af[15] = (__bf16)p3.w;

#pragma unroll
    for (int t = 0; t < 4; ++t)
      acc[t] = __builtin_amdgcn_wmma_f32_16x16x32_bf16(
          false, af, false, bfrag[t], (short)0, acc[t], false, false);
  }

  // ---- Epilogue: bias (+gelu) (+residual), guarded store ----
#pragma unroll
  for (int t = 0; t < 4; ++t) {
    int n = nBase + 16 * t + l16;
#pragma unroll
    for (int r = 0; r < 8; ++r) {
      int m = mBase + r + 8 * hlf;
      if (m < M) {
        float v = acc[t][r] + bias[n];
        if (act) v = gelu_exact(v);
        if (residual) v += residual[(size_t)m * N + n];
        C[(size_t)m * N + n] = v;
      }
    }
  }
}

// -----------------------------------------------------------------------------
// Small multi-head attention: per (batch, head) block; softmax over LK keys.
// Q: (batches, Lq, 256)  K/V: (batches, LK, 256)  O: (batches, Lq, 256)
// -----------------------------------------------------------------------------
template <int LK>
__global__ void attn_kernel(const float* __restrict__ Q,
                            const float* __restrict__ Kv,
                            const float* __restrict__ Vv,
                            float* __restrict__ O, int Lq) {
  const int bh = blockIdx.x;
  const int b = bh >> 3, h = bh & 7;
  const float* qb = Q + (size_t)b * Lq * DM + h * DHH;
  const float* kb = Kv + (size_t)b * LK * DM + h * DHH;
  const float* vb = Vv + (size_t)b * LK * DM + h * DHH;
  const float scale = 0.17677669529663687f;  // 1/sqrt(32)

  for (int l = threadIdx.x; l < Lq; l += blockDim.x) {
    const float* qr = qb + (size_t)l * DM;
    float sc[LK];
    float mx = -3.4e38f;
#pragma unroll
    for (int s = 0; s < LK; ++s) {
      float d = 0.0f;
#pragma unroll
      for (int e = 0; e < DHH; ++e) d += qr[e] * kb[(size_t)s * DM + e];
      d *= scale;
      sc[s] = d;
      mx = fmaxf(mx, d);
    }
    float sum = 0.0f;
#pragma unroll
    for (int s = 0; s < LK; ++s) { sc[s] = expf(sc[s] - mx); sum += sc[s]; }
    const float inv = 1.0f / sum;
    float* orow = O + ((size_t)b * Lq + l) * DM + h * DHH;
#pragma unroll
    for (int e = 0; e < DHH; ++e) {
      float a = 0.0f;
#pragma unroll
      for (int s = 0; s < LK; ++s) a += sc[s] * vb[(size_t)s * DM + e];
      orow[e] = a * inv;
    }
  }
}

// -----------------------------------------------------------------------------
// LayerNorm over last dim (256), one wave per row, in place.
// -----------------------------------------------------------------------------
__global__ void ln_kernel(float* __restrict__ X, const float* __restrict__ g,
                          const float* __restrict__ bta, int M) {
  const int wave = threadIdx.x >> 5, lane = threadIdx.x & 31;
  const int row = blockIdx.x * 8 + wave;
  if (row >= M) return;
  float* xr = X + (size_t)row * DM;
  float v[8];
  float s = 0.0f;
#pragma unroll
  for (int j = 0; j < 8; ++j) { v[j] = xr[lane + 32 * j]; s += v[j]; }
#pragma unroll
  for (int off = 16; off > 0; off >>= 1) s += __shfl_xor(s, off, 32);
  const float mean = s * (1.0f / 256.0f);
  float q = 0.0f;
#pragma unroll
  for (int j = 0; j < 8; ++j) { float d = v[j] - mean; q += d * d; }
#pragma unroll
  for (int off = 16; off > 0; off >>= 1) q += __shfl_xor(q, off, 32);
  const float r = rsqrtf(q * (1.0f / 256.0f) + 1e-5f);
#pragma unroll
  for (int j = 0; j < 8; ++j) {
    int c = lane + 32 * j;
    xr[c] = (v[j] - mean) * r * g[c] + bta[c];
  }
}

// -----------------------------------------------------------------------------
// Elementwise: out = log(relu(in) + 1)
// -----------------------------------------------------------------------------
__global__ void xlog_kernel(const float* __restrict__ in,
                            float* __restrict__ out, size_t n) {
  size_t i = (size_t)blockIdx.x * blockDim.x + threadIdx.x;
  if (i < n) out[i] = logf(fmaxf(in[i], 0.0f) + 1.0f);
}

// (B,V,S,D) row-major -> (B,S,V,D) row-major
__global__ void perm_to_bsvd(const float* __restrict__ in,
                             float* __restrict__ out) {
  size_t idx = (size_t)blockIdx.x * blockDim.x + threadIdx.x;
  const size_t total = (size_t)BB * VV * SS * DM;
  if (idx >= total) return;
  int d = (int)(idx & (DM - 1));
  size_t r = idx >> 8;           // (b*V+v)*S + s
  int s = (int)(r % SS);
  size_t t = r / SS;
  int v = (int)(t % VV);
  int b = (int)(t / VV);
  size_t dst = ((((size_t)b * SS + s) * VV + v) << 8) + d;
  out[dst] = in[idx];
}

// (B,S,V,D) row-major -> (B,V,S,D) row-major
__global__ void perm_to_bvsd(const float* __restrict__ in,
                             float* __restrict__ out) {
  size_t idx = (size_t)blockIdx.x * blockDim.x + threadIdx.x;
  const size_t total = (size_t)BB * VV * SS * DM;
  if (idx >= total) return;
  int d = (int)(idx & (DM - 1));
  size_t r = idx >> 8;           // (b*S+s)*V + v
  int v = (int)(r % VV);
  size_t t = r / VV;
  int s = (int)(t % SS);
  int b = (int)(t / SS);
  size_t dst = ((((size_t)b * VV + v) * SS + s) << 8) + d;
  out[dst] = in[idx];
}

// -----------------------------------------------------------------------------
// SCX fused: per (bs, h, g) row (one wave): scores vs 321 keys, top-33
// selection (tie -> lower index), gather + grouped conv + min-max + exp.
// Qc: (S*G, 256); Kx/Vx: (bs*V, 256); feat: (bs*G, 256)
// -----------------------------------------------------------------------------
__global__ void scx_kernel(const float* __restrict__ Qc,
                           const float* __restrict__ Kx,
                           const float* __restrict__ Vx,
                           const float* __restrict__ cw,
                           const float* __restrict__ cb,
                           float* __restrict__ feat) {
  const int id = blockIdx.x;
  const int g = id % GG;
  const int h = (id / GG) % HH;
  const int bs = id / (GG * HH);
  const int s = bs % SS;
  const int lane = threadIdx.x & 31;
  const float scale = 0.17677669529663687f;

  const float* qr = Qc + (size_t)(s * GG + g) * DM + h * DHH;
  float q[DHH];
#pragma unroll
  for (int e = 0; e < DHH; ++e) q[e] = qr[e];

  float sc[11];
#pragma unroll
  for (int i = 0; i < 11; ++i) {
    int v = lane + 32 * i;
    if (v < VV) {
      const float* kr = Kx + ((size_t)bs * VV + v) * DM + h * DHH;
      float d = 0.0f;
#pragma unroll
      for (int e = 0; e < DHH; ++e) d += q[e] * kr[e];
      sc[i] = d * scale;
    } else {
      sc[i] = -3.4e38f;
    }
  }

  float acc = 0.0f;
  for (int t = 0; t < KK; ++t) {
    // lane-local argmax (lowest index wins ties)
    float bv = -3.4e38f;
    int bi = 0x7fffffff;
#pragma unroll
    for (int i = 0; i < 11; ++i) {
      int v = lane + 32 * i;
      if (sc[i] > bv) { bv = sc[i]; bi = v; }
    }
    // wave argmax butterfly; all lanes converge to the global winner
#pragma unroll
    for (int off = 16; off > 0; off >>= 1) {
      float ov = __shfl_xor(bv, off, 32);
      int oi = __shfl_xor(bi, off, 32);
      if (ov > bv || (ov == bv && oi < bi)) { bv = ov; bi = oi; }
    }
    if ((bi & 31) == lane) sc[bi >> 5] = -3.4e38f;  // owner retires winner
    // grouped conv (kernel-size 1): rank t weighted gather; lane owns d=lane
    acc += cw[g * KK + t] * Vx[((size_t)bs * VV + bi) * DM + h * DHH + lane];
  }
  acc += cb[g];

  float mn = acc, mx = acc;
#pragma unroll
  for (int off = 16; off > 0; off >>= 1) {
    mn = fminf(mn, __shfl_xor(mn, off, 32));
    mx = fmaxf(mx, __shfl_xor(mx, off, 32));
  }
  float xn = expf((acc - mn) / fmaxf(mx - mn, 1e-6f));
  feat[((size_t)bs * GG + g) * DM + h * DHH + lane] = xn;
}

// -----------------------------------------------------------------------------
// Host-side orchestration
// -----------------------------------------------------------------------------
extern "C" void kernel_launch(void* const* d_in, const int* in_sizes, int n_in,
                              void* d_out, int out_size, void* d_ws,
                              size_t ws_size, hipStream_t stream) {
  (void)in_sizes; (void)n_in; (void)out_size; (void)ws_size;

  const float* queries = (const float*)d_in[0];
  const float* t_wq = (const float*)d_in[1];
  const float* t_bq = (const float*)d_in[2];
  const float* t_wk = (const float*)d_in[3];
  const float* t_bk = (const float*)d_in[4];
  const float* t_wv = (const float*)d_in[5];
  const float* t_bv = (const float*)d_in[6];
  const float* t_wo = (const float*)d_in[7];
  const float* t_bo = (const float*)d_in[8];
  const float* s_wq = (const float*)d_in[9];
  const float* s_bq = (const float*)d_in[10];
  const float* s_wk = (const float*)d_in[11];
  const float* s_bk = (const float*)d_in[12];
  const float* s_wv = (const float*)d_in[13];
  const float* s_bv = (const float*)d_in[14];
  const float* s_wo = (const float*)d_in[15];
  const float* s_bo = (const float*)d_in[16];
  const float* cluster = (const float*)d_in[17];
  const float* conv_w = (const float*)d_in[18];
  const float* conv_b = (const float*)d_in[19];
  const float* r_wq = (const float*)d_in[20];
  const float* r_bq = (const float*)d_in[21];
  const float* r_wk = (const float*)d_in[22];
  const float* r_bk = (const float*)d_in[23];
  const float* r_wv = (const float*)d_in[24];
  const float* r_bv = (const float*)d_in[25];
  const float* r_wo = (const float*)d_in[26];
  const float* r_bo = (const float*)d_in[27];
  const float* ln_t1_g = (const float*)d_in[28];
  const float* ln_t1_b = (const float*)d_in[29];
  const float* ln_t2_g = (const float*)d_in[30];
  const float* ln_t2_b = (const float*)d_in[31];
  const float* ln_d1_g = (const float*)d_in[32];
  const float* ln_d1_b = (const float*)d_in[33];
  const float* ln_d2_g = (const float*)d_in[34];
  const float* ln_d2_b = (const float*)d_in[35];
  const float* ft_w1 = (const float*)d_in[36];
  const float* ft_b1 = (const float*)d_in[37];
  const float* ft_w2 = (const float*)d_in[38];
  const float* ft_b2 = (const float*)d_in[39];
  const float* fd_w1 = (const float*)d_in[40];
  const float* fd_b1 = (const float*)d_in[41];
  const float* fd_w2 = (const float*)d_in[42];
  const float* fd_b2 = (const float*)d_in[43];

  const int M1 = BB * VV * SS;             // 61632 tokens
  const size_t MD = (size_t)M1 * DM;
  const int MR = BB * SS * GG;             // 1920 routed rows
  const int MQ = SS * GG;                  // 240 cluster rows

  float* ws = (float*)d_ws;
  float* bufA = ws;                         // M1 x 256
  float* bufQ = bufA + MD;                  // M1 x 256
  float* bufK = bufQ + MD;                  // M1 x 256
  float* bufV = bufK + MD;                  // M1 x 256
  float* bufC = bufV + MD;                  // M1 x 256
  float* bufF = bufC + MD;                  // M1 x 1024
  float* sQc   = bufF + (size_t)M1 * DFFN;  // 240 x 256
  float* sFeat = sQc + (size_t)MQ * DM;     // 1920 x 256
  float* sRoute = sFeat + (size_t)MR * DM;
  float* sRk = sRoute + (size_t)MR * DM;
  float* sRv = sRk + (size_t)MR * DM;

  // bf16 pre-swizzled weights (32-byte aligned: all prior sizes are x256 floats)
  __bf16* wp = (__bf16*)(sRv + (size_t)MR * DM);
  size_t wo = 0;
  const size_t SZ_DD = (size_t)DM * DM;       // 65536
  const size_t SZ_DF = (size_t)DM * DFFN;     // 262144
  __bf16* p_twq = wp + wo; wo += SZ_DD;
  __bf16* p_twk = wp + wo; wo += SZ_DD;
  __bf16* p_twv = wp + wo; wo += SZ_DD;
  __bf16* p_two = wp + wo; wo += SZ_DD;
  __bf16* p_swq = wp + wo; wo += SZ_DD;
  __bf16* p_swk = wp + wo; wo += SZ_DD;
  __bf16* p_swv = wp + wo; wo += SZ_DD;
  __bf16* p_swo = wp + wo; wo += SZ_DD;
  __bf16* p_rwq = wp + wo; wo += SZ_DD;
  __bf16* p_rwk = wp + wo; wo += SZ_DD;
  __bf16* p_rwv = wp + wo; wo += SZ_DD;
  __bf16* p_rwo = wp + wo; wo += SZ_DD;
  __bf16* p_fw1 = wp + wo; wo += SZ_DF;
  __bf16* p_fw2 = wp + wo; wo += SZ_DF;
  __bf16* p_gw1 = wp + wo; wo += SZ_DF;
  __bf16* p_gw2 = wp + wo; wo += SZ_DF;

  const dim3 blk(256);
  auto gg = [](int M, int N) { return dim3((N + 63) / 64, (M + 127) / 128); };
  const dim3 eltGrid((unsigned)((MD + 255) / 256));
  const dim3 wgridDD((unsigned)(SZ_DD / 256));
  const dim3 wgridDF((unsigned)(SZ_DF / 256));

  // ---- weight conversion / swizzle (once per launch, ~7 MB total) ----
  wconv_kernel<<<wgridDD, blk, 0, stream>>>(t_wq, p_twq, DM, DM);
  wconv_kernel<<<wgridDD, blk, 0, stream>>>(t_wk, p_twk, DM, DM);
  wconv_kernel<<<wgridDD, blk, 0, stream>>>(t_wv, p_twv, DM, DM);
  wconv_kernel<<<wgridDD, blk, 0, stream>>>(t_wo, p_two, DM, DM);
  wconv_kernel<<<wgridDD, blk, 0, stream>>>(s_wq, p_swq, DM, DM);
  wconv_kernel<<<wgridDD, blk, 0, stream>>>(s_wk, p_swk, DM, DM);
  wconv_kernel<<<wgridDD, blk, 0, stream>>>(s_wv, p_swv, DM, DM);
  wconv_kernel<<<wgridDD, blk, 0, stream>>>(s_wo, p_swo, DM, DM);
  wconv_kernel<<<wgridDD, blk, 0, stream>>>(r_wq, p_rwq, DM, DM);
  wconv_kernel<<<wgridDD, blk, 0, stream>>>(r_wk, p_rwk, DM, DM);
  wconv_kernel<<<wgridDD, blk, 0, stream>>>(r_wv, p_rwv, DM, DM);
  wconv_kernel<<<wgridDD, blk, 0, stream>>>(r_wo, p_rwo, DM, DM);
  wconv_kernel<<<wgridDF, blk, 0, stream>>>(ft_w1, p_fw1, DM, DFFN);
  wconv_kernel<<<wgridDF, blk, 0, stream>>>(ft_w2, p_fw2, DFFN, DM);
  wconv_kernel<<<wgridDF, blk, 0, stream>>>(fd_w1, p_gw1, DM, DFFN);
  wconv_kernel<<<wgridDF, blk, 0, stream>>>(fd_w2, p_gw2, DFFN, DM);

  // ===================== Route 1: temporal =====================
  gemm_bf16_wmma<<<gg(M1, DM), blk, 0, stream>>>(queries, p_twq, t_bq, nullptr, bufQ, M1, DM, DM, 0);
  gemm_bf16_wmma<<<gg(M1, DM), blk, 0, stream>>>(queries, p_twk, t_bk, nullptr, bufK, M1, DM, DM, 0);
  gemm_bf16_wmma<<<gg(M1, DM), blk, 0, stream>>>(queries, p_twv, t_bv, nullptr, bufV, M1, DM, DM, 0);
  attn_kernel<SS><<<dim3(BB * VV * HH), dim3(32), 0, stream>>>(bufQ, bufK, bufV, bufC, SS);
  gemm_bf16_wmma<<<gg(M1, DM), blk, 0, stream>>>(bufC, p_two, t_bo, queries, bufA, M1, DM, DM, 0);
  ln_kernel<<<dim3((M1 + 7) / 8), blk, 0, stream>>>(bufA, ln_t1_g, ln_t1_b, M1);
  gemm_bf16_wmma<<<gg(M1, DFFN), blk, 0, stream>>>(bufA, p_fw1, ft_b1, nullptr, bufF, M1, DM, DFFN, 1);
  gemm_bf16_wmma<<<gg(M1, DM), blk, 0, stream>>>(bufF, p_fw2, ft_b2, bufA, bufC, M1, DFFN, DM, 0);
  ln_kernel<<<dim3((M1 + 7) / 8), blk, 0, stream>>>(bufC, ln_t2_g, ln_t2_b, M1);

  // ===================== Route 2: dimensional =====================
  perm_to_bsvd<<<eltGrid, blk, 0, stream>>>(bufC, bufQ);              // XD = bufQ
  xlog_kernel<<<eltGrid, blk, 0, stream>>>(bufQ, bufK, MD);           // XL = bufK
  gemm_bf16_wmma<<<gg(M1, DM), blk, 0, stream>>>(bufK, p_swk, s_bk, nullptr, bufV, M1, DM, DM, 0);  // Kx
  gemm_bf16_wmma<<<gg(M1, DM), blk, 0, stream>>>(bufK, p_swv, s_bv, nullptr, bufA, M1, DM, DM, 0);  // Vx
  gemm_bf16_wmma<<<gg(MQ, DM), blk, 0, stream>>>(cluster, p_swq, s_bq, nullptr, sQc, MQ, DM, DM, 0);
  scx_kernel<<<dim3(BB * SS * HH * GG), dim3(32), 0, stream>>>(sQc, bufV, bufA, conv_w, conv_b, sFeat);
  gemm_bf16_wmma<<<gg(MR, DM), blk, 0, stream>>>(sFeat, p_swo, s_bo, nullptr, sRoute, MR, DM, DM, 0);

  gemm_bf16_wmma<<<gg(M1, DM), blk, 0, stream>>>(bufQ, p_rwq, r_bq, nullptr, bufK, M1, DM, DM, 0);  // rq
  gemm_bf16_wmma<<<gg(MR, DM), blk, 0, stream>>>(sRoute, p_rwk, r_bk, nullptr, sRk, MR, DM, DM, 0);
  gemm_bf16_wmma<<<gg(MR, DM), blk, 0, stream>>>(sRoute, p_rwv, r_bv, nullptr, sRv, MR, DM, DM, 0);
  attn_kernel<GG><<<dim3(BB * SS * HH), dim3(128), 0, stream>>>(bufK, sRk, sRv, bufC, VV);
  gemm_bf16_wmma<<<gg(M1, DM), blk, 0, stream>>>(bufC, p_rwo, r_bo, bufQ, bufV, M1, DM, DM, 0);
  ln_kernel<<<dim3((M1 + 7) / 8), blk, 0, stream>>>(bufV, ln_d1_g, ln_d1_b, M1);
  gemm_bf16_wmma<<<gg(M1, DFFN), blk, 0, stream>>>(bufV, p_gw1, fd_b1, nullptr, bufF, M1, DM, DFFN, 1);
  gemm_bf16_wmma<<<gg(M1, DM), blk, 0, stream>>>(bufF, p_gw2, fd_b2, bufV, bufA, M1, DFFN, DM, 0);
  ln_kernel<<<dim3((M1 + 7) / 8), blk, 0, stream>>>(bufA, ln_d2_g, ln_d2_b, M1);

  // Final layout: (B,S,V,D) -> (B,V,S,D)
  perm_to_bvsd<<<eltGrid, blk, 0, stream>>>(bufA, (float*)d_out);
}
